// PredicateHead_63969242907161
// MI455X (gfx1250) — matrix operand order repaired and checked by reading, not compiled, not run on hardware
//
#include <hip/hip_runtime.h>
#include <hip/hip_bf16.h>

typedef __bf16 bf16;
typedef __attribute__((ext_vector_type(16))) __bf16 v16bf;
typedef __attribute__((ext_vector_type(8)))  float  v8f;
typedef __attribute__((ext_vector_type(4)))  unsigned int v4u;
typedef __attribute__((ext_vector_type(8)))  int v8i;
typedef __attribute__((ext_vector_type(4)))  int v4i;

#define NB 4096   // batch
#define ND 1024   // input dim
#define NH 256    // hidden 1
#define NM 128    // hidden 2 (H/2)
#define NK 64     // predicate heads
#define BT 64     // batch rows per workgroup
#define KT 32     // K-step per WMMA (bf16 16x16x32)

union FragBF { v16bf v; unsigned int u[8]; };

__device__ __forceinline__ float silu_f(float x) {
  return x / (1.0f + __expf(-x));
}
__device__ __forceinline__ bf16 to_bf16(float x) { return (bf16)x; }
__device__ __forceinline__ bf16 to_bf16(bf16 x)  { return x; }

// ---- Tensor Data Mover: 2-D bf16 tile (rows x cols) from global -> LDS ----
// Descriptor packed per CDNA5 ISA sec 8.3/8.4 (group0 128b, group1 256b).
__device__ __forceinline__ void tdm_load_2d_bf16(unsigned lds_off, const void* gptr,
                                                 unsigned rows, unsigned cols,
                                                 unsigned row_stride_elems) {
  const unsigned long long ga = (unsigned long long)(uintptr_t)gptr;
  v4u g0;
  g0[0] = 1u;                                             // count=1, user D#
  g0[1] = lds_off;                                        // lds_addr (bytes)
  g0[2] = (unsigned)ga;                                   // global_addr[31:0]
  g0[3] = ((unsigned)(ga >> 32) & 0x01FFFFFFu)            // global_addr[56:32]
          | 0x80000000u;                                  // type=2 ("image")
  v8i g1;
  g1[0] = (int)(1u << 16);                                // data_size=2B; no mask/pad
  g1[1] = (int)((cols & 0xFFFFu) << 16);                  // tensor_dim0[15:0]
  g1[2] = (int)(((cols >> 16) & 0xFFFFu)                  // tensor_dim0[31:16]
          | ((rows & 0xFFFFu) << 16));                    // tensor_dim1[15:0]
  g1[3] = (int)(((rows >> 16) & 0xFFFFu)                  // tensor_dim1[31:16]
          | ((cols & 0xFFFFu) << 16));                    // tile_dim0
  g1[4] = (int)(rows & 0xFFFFu);                          // tile_dim1 (tile_dim2=0)
  g1[5] = (int)row_stride_elems;                          // tensor_dim0_stride[31:0]
  g1[6] = 0;                                              // stride hi / dim1_stride lo
  g1[7] = 0;
  v4i z4 = {0, 0, 0, 0};
#if defined(__clang_major__) && (__clang_major__ >= 23)
  v8i z8 = {0, 0, 0, 0, 0, 0, 0, 0};
  __builtin_amdgcn_tensor_load_to_lds(g0, g1, z4, z4, z8, 0);
#else
  __builtin_amdgcn_tensor_load_to_lds(g0, g1, z4, z4, 0);
#endif
}

// ---- f32 -> bf16 conversion pre-pass (4 elements / thread) ----
__global__ __launch_bounds__(256)
void cvt_f32_to_bf16(const float* __restrict__ src, bf16* __restrict__ dst, int n4) {
  const int i = blockIdx.x * 256 + threadIdx.x;
  if (i < n4) {
    const float4 v = ((const float4*)src)[i];
    bf16 o[4] = {(bf16)v.x, (bf16)v.y, (bf16)v.z, (bf16)v.w};
    *(unsigned long long*)(dst + (size_t)i * 4) = *(unsigned long long*)o;
  }
}

// ---- Fused 3-layer predicate heads.  T = float (convert in-kernel) or bf16
// (pre-converted; A-tile staged via TDM tensor_load_to_lds). ----
template <typename T, bool USE_TDM>
__global__ __launch_bounds__(256)
void predicate_heads_fused(
    const T*     __restrict__ pooledX,     // [B, D]
    const T*     __restrict__ W1X,         // [K, D, H]
    const T*     __restrict__ W2X,         // [K, H, M]
    const int*   __restrict__ sport_ids,   // [B]
    const float* __restrict__ b1,          // [K, H]
    const float* __restrict__ b2,          // [K, M]
    const float* __restrict__ W3,          // [K, M]
    const float* __restrict__ b3,          // [K]
    const unsigned char* __restrict__ mask_table, // [3, K] bool
    float* __restrict__ out_logits,        // [B, K]
    float* __restrict__ out_probs)         // [B, K]
{
  __shared__ bf16 ldsA[BT * KT];       // pooled tile [row][kk]              (4 KB)
  __shared__ bf16 ldsB[NH * KT];       // W1 tile transposed [h][kk]         (16 KB)
  __shared__ bf16 ldsW2T[NM * NH];     // W2[k] transposed [m][h]            (64 KB)
  __shared__ bf16 ldsH1[BT * NH];      // silu(h1) tile [row][h]             (32 KB)
  __shared__ float ldsLogits[BT];

  const int tid  = threadIdx.x;
  const int lane = tid & 31;
  const int wave = tid >> 5;
  const int half = lane >> 4;
  const int lx   = lane & 15;
  const int rowBlk  = wave & 3;
  const int colHalf = wave >> 2;
  const int kp = blockIdx.x;
  const int b0 = blockIdx.y * BT;

  if (tid < BT) ldsLogits[tid] = 0.0f;

  // ---- Stage W2[kp] transposed into LDS once ----
  {
    const T* w2row = W2X + ((size_t)kp * NH + tid) * NM;
    #pragma unroll 4
    for (int m = 0; m < NM; ++m)
      ldsW2T[m * NH + tid] = to_bf16(w2row[m]);
  }

  // ---- Layer 1: 64x256 tile of silu(pooled @ W1[kp] + b1) ----
  v8f acc1[8];
  #pragma unroll
  for (int t = 0; t < 8; ++t)
    #pragma unroll
    for (int j = 0; j < 8; ++j) acc1[t][j] = 0.0f;

  const int arow = tid >> 2;
  const int aseg = (tid & 3) * 8;
  const int br   = tid & 31;
  const int bcg  = tid >> 5;
  const unsigned ldsA_off = (unsigned)(uintptr_t)&ldsA[0];

  for (int d0 = 0; d0 < ND; d0 += KT) {
    __syncthreads();
    if constexpr (USE_TDM) {
      // Async DMA of the 64x32 bf16 A-tile straight into LDS (TENSORcnt).
      if (wave == 0)
        tdm_load_2d_bf16(ldsA_off, pooledX + (size_t)b0 * ND + d0, BT, KT, ND);
    } else {
      const T* src = pooledX + (size_t)(b0 + arow) * ND + d0 + aseg;
      #pragma unroll
      for (int j = 0; j < 8; ++j)
        ldsA[arow * KT + aseg + j] = to_bf16(src[j]);
    }
    // B tile transposed: W1[kp][d0+br][h] -> ldsB[h][br]
    {
      const T* src = W1X + ((size_t)kp * ND + d0 + br) * NH + bcg * KT;
      if (d0 + KT < ND) __builtin_prefetch(src + (size_t)KT * NH, 0, 1);
      #pragma unroll 8
      for (int j = 0; j < KT; ++j)
        ldsB[(bcg * KT + j) * KT + br] = to_bf16(src[j]);
    }
    if constexpr (USE_TDM) {
      if (wave == 0) __builtin_amdgcn_s_wait_tensorcnt(0);
    }
    __syncthreads();

    // A fragment (16x32 bf16): two 16B chunks per lane (contiguous k = 16*half + s)
    FragBF a;
    {
      const unsigned int* p0 =
        (const unsigned int*)&ldsA[(rowBlk * 16 + lx) * KT + half * 16];
      const unsigned int* p1 =
        (const unsigned int*)&ldsA[(rowBlk * 16 + lx) * KT + half * 16 + 8];
      #pragma unroll
      for (int j = 0; j < 4; ++j) { a.u[j] = p0[j]; a.u[4 + j] = p1[j]; }
    }
    #pragma unroll
    for (int t = 0; t < 8; ++t) {
      const int n = colHalf * 128 + t * 16 + lx;
      FragBF b;
      const unsigned int* p =
        (const unsigned int*)&ldsB[n * KT + half * 16];
      #pragma unroll
      for (int j = 0; j < 8; ++j) b.u[j] = p[j];
      acc1[t] = __builtin_amdgcn_wmma_f32_16x16x32_bf16(
          false, a.v, false, b.v, (short)0, acc1[t], false, false);
    }
  }

  // bias + silu -> ldsH1 (bf16)
  __syncthreads();
  #pragma unroll
  for (int t = 0; t < 8; ++t) {
    const int n = colHalf * 128 + t * 16 + lx;
    const float bias = b1[kp * NH + n];
    #pragma unroll
    for (int j = 0; j < 8; ++j) {
      const int row = rowBlk * 16 + half * 8 + j;   // C/D layout: M = j + 8*half
      ldsH1[row * NH + n] = (bf16)silu_f(acc1[t][j] + bias);
    }
  }
  __syncthreads();

  // ---- Layer 2: 64x128 tile of silu(h1 @ W2[kp] + b2) ----
  v8f acc2[4];
  #pragma unroll
  for (int t = 0; t < 4; ++t)
    #pragma unroll
    for (int j = 0; j < 8; ++j) acc2[t][j] = 0.0f;

  for (int h0 = 0; h0 < NH; h0 += KT) {
    FragBF a;
    {
      const unsigned int* p0 =
        (const unsigned int*)&ldsH1[(rowBlk * 16 + lx) * NH + h0 + half * 16];
      const unsigned int* p1 =
        (const unsigned int*)&ldsH1[(rowBlk * 16 + lx) * NH + h0 + half * 16 + 8];
      #pragma unroll
      for (int j = 0; j < 4; ++j) { a.u[j] = p0[j]; a.u[4 + j] = p1[j]; }
    }
    #pragma unroll
    for (int t = 0; t < 4; ++t) {
      const int n = colHalf * 64 + t * 16 + lx;
      FragBF b;
      const unsigned int* p =
        (const unsigned int*)&ldsW2T[n * NH + h0 + half * 16];
      #pragma unroll
      for (int j = 0; j < 8; ++j) b.u[j] = p[j];
      acc2[t] = __builtin_amdgcn_wmma_f32_16x16x32_bf16(
          false, a.v, false, b.v, (short)0, acc2[t], false, false);
    }
  }

  // ---- Layer 3: logits = h2 . W3[kp] via LDS float atomics ----
  #pragma unroll
  for (int t = 0; t < 4; ++t) {
    const int n = colHalf * 64 + t * 16 + lx;
    const float bias = b2[kp * NM + n];
    const float w3   = W3[kp * NM + n];
    #pragma unroll
    for (int j = 0; j < 8; ++j) {
      const int row = rowBlk * 16 + half * 8 + j;
      atomicAdd(&ldsLogits[row], silu_f(acc2[t][j] + bias) * w3);
    }
  }
  __syncthreads();

  // ---- Tail: bias, sport mask, sigmoid ----
  if (tid < BT) {
    const int b = b0 + tid;
    float logit = ldsLogits[tid] + b3[kp];
    const int sport = sport_ids[b];
    if (!mask_table[sport * NK + kp]) logit = -INFINITY;
    const float prob = 1.0f / (1.0f + __expf(-logit));
    out_logits[(size_t)b * NK + kp] = logit;
    out_probs[(size_t)b * NK + kp]  = prob;
  }
}

extern "C" void kernel_launch(void* const* d_in, const int* in_sizes, int n_in,
                              void* d_out, int out_size, void* d_ws, size_t ws_size,
                              hipStream_t stream) {
  (void)in_sizes; (void)n_in; (void)out_size;
  const float* pooled          = (const float*)d_in[0];
  const int*   sport_ids       = (const int*)d_in[1];
  const float* W1              = (const float*)d_in[2];
  const float* b1              = (const float*)d_in[3];
  const float* W2              = (const float*)d_in[4];
  const float* b2              = (const float*)d_in[5];
  const float* W3              = (const float*)d_in[6];
  const float* b3              = (const float*)d_in[7];
  const unsigned char* mask_tb = (const unsigned char*)d_in[8];

  float* out        = (float*)d_out;
  float* out_logits = out;
  float* out_probs  = out + (size_t)NB * NK;

  const size_t nPooled = (size_t)NB * ND;
  const size_t nW1     = (size_t)NK * ND * NH;
  const size_t nW2     = (size_t)NK * NH * NM;
  const size_t need    = (nPooled + nW1 + nW2) * sizeof(bf16);

  dim3 grid(NK, NB / BT);   // 64 heads x 64 batch tiles

  if (ws_size >= need) {
    // bf16 fast path: one-shot conversion into workspace, then TDM-fed WMMA.
    bf16* pb  = (bf16*)d_ws;
    bf16* w1b = pb + nPooled;
    bf16* w2b = w1b + nW1;
    int n4;
    n4 = (int)(nPooled / 4);
    cvt_f32_to_bf16<<<(n4 + 255) / 256, 256, 0, stream>>>(pooled, pb, n4);
    n4 = (int)(nW1 / 4);
    cvt_f32_to_bf16<<<(n4 + 255) / 256, 256, 0, stream>>>(W1, w1b, n4);
    n4 = (int)(nW2 / 4);
    cvt_f32_to_bf16<<<(n4 + 255) / 256, 256, 0, stream>>>(W2, w2b, n4);
    predicate_heads_fused<bf16, true><<<grid, 256, 0, stream>>>(
        pb, w1b, w2b, sport_ids, b1, b2, W3, b3, mask_tb,
        out_logits, out_probs);
  } else {
    // Fallback: convert-on-the-fly from f32 inputs.
    predicate_heads_fused<float, false><<<grid, 256, 0, stream>>>(
        pooled, W1, W2, sport_ids, b1, b2, W3, b3, mask_tb,
        out_logits, out_probs);
  }
}
